// ScaledCosineSelfAttention_73272142069949
// MI455X (gfx1250) — compile-verified, matrix operand order checked by reading, not run on hardware
//
#include <hip/hip_runtime.h>
#include <stdint.h>

typedef unsigned short u16;
typedef __attribute__((ext_vector_type(16))) __bf16 v16bf;
typedef __attribute__((ext_vector_type(8)))  float  v8f;
typedef __attribute__((ext_vector_type(4)))  unsigned v4u;
typedef __attribute__((ext_vector_type(8)))  int      v8i;
typedef __attribute__((ext_vector_type(4)))  int      v4i;

#define HB 2
#define HT 2048
#define HC 2048
#define HH 16
#define HD 128
#define HM (HB*HT)   // 4096 rows for the projection GEMMs

#if defined(__gfx1250__) && __has_builtin(__builtin_amdgcn_tensor_load_to_lds) && \
    __has_builtin(__builtin_amdgcn_s_wait_tensorcnt)
#define USE_TDM 1
#else
#define USE_TDM 0
#endif

__device__ __forceinline__ u16 f2bf(float f) {
  unsigned u = __float_as_uint(f);
  u += 0x7FFFu + ((u >> 16) & 1u);   // round-to-nearest-even
  return (u16)(u >> 16);
}
__device__ __forceinline__ float bf2f(u16 b) {
  return __uint_as_float(((unsigned)b) << 16);
}

__device__ __forceinline__ v8f wmma_bf16(v16bf a, v16bf b, v8f c) {
  // D = A(16x32 bf16) * B(32x16 bf16) + C(16x16 f32)
  return __builtin_amdgcn_wmma_f32_16x16x32_bf16(false, a, false, b, (short)0, c,
                                                 false, false);
}

#if USE_TDM
// Issue TENSOR_LOAD_TO_LDS of a 2-D bf16 tile: tile_d0 elems along dim0
// (contiguous), tile_d1 rows, row stride d0_stride elems. D# packing per
// CDNA5 ISA (group0 128b, group1 256b; groups 2/3 and the
// trailing group are zero for a 2-D tensor). 6-arg clang-23 builtin form.
__device__ __forceinline__ void tdm_load_tile_2d(unsigned lds_byte_addr,
                                                 const void* gptr,
                                                 unsigned tensor_d0,
                                                 unsigned tensor_d1,
                                                 unsigned tile_d0,
                                                 unsigned tile_d1,
                                                 unsigned d0_stride) {
  unsigned long long ga = (unsigned long long)(uintptr_t)gptr;
  v4u g0;
  g0[0] = 1u;                                   // count=1 (valid), user mode
  g0[1] = lds_byte_addr;                        // lds_addr
  g0[2] = (unsigned)ga;                         // global_addr[31:0]
  g0[3] = (unsigned)((ga >> 32) & 0x01FFFFFFull) | (2u << 30); // addr[56:32], type=2
  v8i g1;
  g1[0] = (int)(1u << 16);                      // wg_mask=0, data_size=1 (2 bytes)
  g1[1] = (int)((tensor_d0 & 0xFFFFu) << 16);   // tensor_dim0[15:0] @ bits 63:48
  g1[2] = (int)(((tensor_d0 >> 16) & 0xFFFFu) | ((tensor_d1 & 0xFFFFu) << 16));
  g1[3] = (int)(((tensor_d1 >> 16) & 0xFFFFu) | ((tile_d0 & 0xFFFFu) << 16));
  g1[4] = (int)(tile_d1 & 0xFFFFu);             // tile_dim1, tile_dim2=0
  g1[5] = (int)d0_stride;                       // tensor_dim0_stride[31:0]
  g1[6] = 0;                                    // stride hi, dim1_stride lo
  g1[7] = 0;
  v4i z4 = {0, 0, 0, 0};
  v8i z8 = {0, 0, 0, 0, 0, 0, 0, 0};
  __builtin_amdgcn_tensor_load_to_lds(g0, g1, z4, z4, z8, 0);
}
#endif

// ---------------------------------------------------------------- convert
__global__ void f32_to_bf16_kernel(const float* __restrict__ in,
                                   u16* __restrict__ out, long n) {
  long i = (long)blockIdx.x * blockDim.x + threadIdx.x;
  if (i < n) out[i] = f2bf(in[i]);
}

// ---------------------------------------------------------------- GEMM NT
// C[M,N] = A[M,K] * B[N,K]^T; TDM double-buffered LDS staging on gfx1250.
__global__ __launch_bounds__(256) void gemm_nt_bf16(const u16* __restrict__ A,
                                                    const u16* __restrict__ Bw,
                                                    float* __restrict__ Cout,
                                                    int M, int N, int K) {
  __shared__ u16 As[2][128 * 32];
  __shared__ u16 Bs[2][128 * 32];

  const int tid  = threadIdx.x;
  const int lane = tid & 31;
  const int wave = tid >> 5;          // 8 waves
  const int wm   = (wave >> 2) * 64;
  const int wn   = (wave & 3) * 32;
  const int blockM = blockIdx.y * 128;
  const int blockN = blockIdx.x * 128;
  const int lr = lane & 15;
  const int lh = (lane >> 4) << 4;
  const int hi = lane >> 4;

  v8f acc[4][2];
  const v8f zacc = {};
  #pragma unroll
  for (int f = 0; f < 4; ++f)
    #pragma unroll
    for (int g = 0; g < 2; ++g) acc[f][g] = zacc;

#if USE_TDM
  const unsigned asAddr[2] = { (unsigned)(uintptr_t)&As[0][0],
                               (unsigned)(uintptr_t)&As[1][0] };
  const unsigned bsAddr[2] = { (unsigned)(uintptr_t)&Bs[0][0],
                               (unsigned)(uintptr_t)&Bs[1][0] };
  if (tid < 32) {   // wave 0 drives the TDM pipeline
    tdm_load_tile_2d(asAddr[0], A  + (size_t)blockM * K, K, M, 32, 128, K);
    tdm_load_tile_2d(bsAddr[0], Bw + (size_t)blockN * K, K, N, 32, 128, K);
  }
#else
  const int row = tid >> 1;
  const int kc  = (tid & 1) << 4;
#endif

  int cur = 0;
  for (int kt = 0; kt < K; kt += 32, cur ^= 1) {
#if USE_TDM
    if (tid < 32) {
      const int nxt = kt + 32;
      if (nxt < K) {
        // prefetch next tile pair into the other buffer (readers of it
        // finished at the previous iteration's trailing barrier)
        tdm_load_tile_2d(asAddr[cur ^ 1], A  + (size_t)blockM * K + nxt,
                         K, M, 32, 128, K);
        tdm_load_tile_2d(bsAddr[cur ^ 1], Bw + (size_t)blockN * K + nxt,
                         K, N, 32, 128, K);
        __builtin_amdgcn_s_wait_tensorcnt(2);  // pair for buf[cur] done (in-order)
      } else {
        __builtin_amdgcn_s_wait_tensorcnt(0);
      }
    }
    __syncthreads();                            // buf[cur] visible to all waves
#else
    {
      const uint4* ga = (const uint4*)(A  + (size_t)(blockM + row) * K + kt + kc);
      const uint4* gb = (const uint4*)(Bw + (size_t)(blockN + row) * K + kt + kc);
      uint4 a0 = ga[0], a1 = ga[1];
      uint4 b0 = gb[0], b1 = gb[1];
      __syncthreads();
      *(uint4*)&As[cur][row * 32 + kc]     = a0;
      *(uint4*)&As[cur][row * 32 + kc + 8] = a1;
      *(uint4*)&Bs[cur][row * 32 + kc]     = b0;
      *(uint4*)&Bs[cur][row * 32 + kc + 8] = b1;
      __syncthreads();
    }
#endif

    v16bf af[4], bfr[2];
    #pragma unroll
    for (int f = 0; f < 4; ++f)
      af[f] = *(const v16bf*)&As[cur][(wm + f * 16 + lr) * 32 + lh];
    #pragma unroll
    for (int g = 0; g < 2; ++g)
      bfr[g] = *(const v16bf*)&Bs[cur][(wn + g * 16 + lr) * 32 + lh];
    #pragma unroll
    for (int f = 0; f < 4; ++f)
      #pragma unroll
      for (int g = 0; g < 2; ++g)
        acc[f][g] = wmma_bf16(af[f], bfr[g], acc[f][g]);

    __syncthreads();                            // all reads of buf[cur] done
  }

  #pragma unroll
  for (int f = 0; f < 4; ++f)
    #pragma unroll
    for (int g = 0; g < 2; ++g)
      #pragma unroll
      for (int r = 0; r < 8; ++r) {
        int rr = blockM + wm + f * 16 + r + 8 * hi;
        int cc = blockN + wn + g * 16 + lr;
        Cout[(size_t)rr * N + cc] = acc[f][g][r];
      }
}

// ------------------------------------------------------- l2norm + relayout
// in:  fp32 [B,T,H,D] (projection output), out: bf16 [B,H,T,D]
__global__ __launch_bounds__(128) void l2norm_bf16_kernel(const float* __restrict__ in,
                                                          u16* __restrict__ out) {
  const int rid = blockIdx.x;     // (b*T + t)*H + h
  const int tid = threadIdx.x;    // d
  float v  = in[(size_t)rid * HD + tid];
  float ss = v * v;
  #pragma unroll
  for (int off = 16; off > 0; off >>= 1) ss += __shfl_xor(ss, off, 32);
  __shared__ float red[4];
  if ((tid & 31) == 0) red[tid >> 5] = ss;
  __syncthreads();
  float tot = red[0] + red[1] + red[2] + red[3];
  float inv = 1.f / (sqrtf(tot) + 1e-8f);
  int b = rid / (HT * HH);
  int t = (rid / HH) % HT;
  int h = rid % HH;
  size_t oo = (((size_t)(b * HH + h)) * HT + t) * HD + tid;
  out[oo] = f2bf(v * inv);
}

// ------------------------------------------------------- flash attention
// One block = one (b,h) pair, 64 query rows. Causal, online softmax.
__global__ __launch_bounds__(256) void attn_kernel(const u16* __restrict__ Qb,
                                                   const u16* __restrict__ Kb,
                                                   const u16* __restrict__ Vb,
                                                   float* __restrict__ out,
                                                   const float* __restrict__ taup,
                                                   const float* __restrict__ nup) {
  __shared__ u16 Qs[64 * 128];    // [q][d]
  __shared__ u16 Ks[64 * 128];    // [k][d]
  __shared__ u16 Vs[128 * 64];    // transposed: [d][k]
  __shared__ u16 SP[64 * 64];     // scores (bf16), overwritten in place by P
  __shared__ float m_s[64], l_s[64], sc_s[64];

  const int tid  = threadIdx.x;
  const int lane = tid & 31;
  const int wave = tid >> 5;
  const int lr = lane & 15;
  const int lh = (lane >> 4) << 4;
  const int hi = lane >> 4;
  const int bh = blockIdx.y;
  const int b = bh >> 4, h = bh & 15;
  const int q0 = blockIdx.x * 64;
  const float tauv = taup[0];
  const float nuv  = nup[0];

  // Q tile is contiguous in [B,H,T,D]
  {
    const uint4* gq = (const uint4*)(Qb + ((size_t)bh * HT + q0) * HD);
    #pragma unroll
    for (int i = 0; i < 4; ++i)
      ((uint4*)Qs)[tid + i * 256] = gq[tid + i * 256];
  }
  if (tid < 64) { m_s[tid] = -1e30f; l_s[tid] = 0.f; }

  const v8f zacc = {};
  v8f o[4];
  #pragma unroll
  for (int c = 0; c < 4; ++c) o[c] = zacc;

  const int tm  = wave >> 1;        // S/O row tile (0..3)
  const int tnb = (wave & 1) * 2;   // S col tile base
  const int onb = (wave & 1) * 4;   // O col tile base

  for (int jt = 0; jt <= (int)blockIdx.x; ++jt) {
    const int k0 = jt * 64;
    // prefetch next K/V tiles into cache while this iteration runs
    if (jt + 1 <= (int)blockIdx.x) {
      const u16* nk = Kb + ((size_t)bh * HT + k0 + 64) * HD;
      const u16* nv = Vb + ((size_t)bh * HT + k0 + 64) * HD;
      __builtin_prefetch(nk + tid * 32, 0, 1);
      __builtin_prefetch(nv + tid * 32, 0, 1);
    }
    __syncthreads();
    {  // stage K tile (contiguous)
      const uint4* gk = (const uint4*)(Kb + ((size_t)bh * HT + k0) * HD);
      #pragma unroll
      for (int i = 0; i < 4; ++i)
        ((uint4*)Ks)[tid + i * 256] = gk[tid + i * 256];
    }
    {  // stage V tile transposed -> [d][k]
      const u16* gv = Vb + ((size_t)bh * HT + k0) * HD;
      int r  = tid >> 2;
      int d0 = (tid & 3) * 32;
      #pragma unroll
      for (int p = 0; p < 4; ++p) {
        uint4 u = ((const uint4*)(gv + (size_t)r * HD + d0))[p];
        const u16* us = (const u16*)&u;
        #pragma unroll
        for (int e = 0; e < 8; ++e)
          Vs[(d0 + p * 8 + e) * 64 + r] = us[e];
      }
    }
    __syncthreads();

    // S = Q K^T  (K-dim = 128 -> 4 WMMA steps)
    v8f s[2];
    s[0] = zacc; s[1] = zacc;
    #pragma unroll
    for (int kk = 0; kk < 128; kk += 32) {
      v16bf aq = *(const v16bf*)&Qs[(tm * 16 + lr) * HD + kk + lh];
      #pragma unroll
      for (int g = 0; g < 2; ++g) {
        v16bf bk = *(const v16bf*)&Ks[((tnb + g) * 16 + lr) * HD + kk + lh];
        s[g] = wmma_bf16(aq, bk, s[g]);
      }
    }
    // tau scale + causal mask, store bf16 scores
    #pragma unroll
    for (int g = 0; g < 2; ++g)
      #pragma unroll
      for (int r = 0; r < 8; ++r) {
        int srow = tm * 16 + r + 8 * hi;
        int scol = (tnb + g) * 16 + lr;
        float v = s[g][r] * tauv;
        if (k0 + scol > q0 + srow) v = -1e30f;
        SP[srow * 64 + scol] = f2bf(v);
      }
    __syncthreads();

    // online softmax: 4 threads per query row, P written in place (bf16)
    {
      int rowq = tid >> 2;
      int sub  = tid & 3;
      float mloc = -1e30f;
      #pragma unroll
      for (int i = 0; i < 16; ++i)
        mloc = fmaxf(mloc, bf2f(SP[rowq * 64 + sub * 16 + i]));
      mloc = fmaxf(mloc, __shfl_xor(mloc, 1, 32));
      mloc = fmaxf(mloc, __shfl_xor(mloc, 2, 32));
      float mold = m_s[rowq];
      float mnew = fmaxf(mold, mloc);
      float lloc = 0.f;
      #pragma unroll
      for (int i = 0; i < 16; ++i) {
        float e = __expf(bf2f(SP[rowq * 64 + sub * 16 + i]) - mnew);
        lloc += e;
        SP[rowq * 64 + sub * 16 + i] = f2bf(e);
      }
      lloc += __shfl_xor(lloc, 1, 32);
      lloc += __shfl_xor(lloc, 2, 32);
      if (sub == 0) {
        float sc = __expf(mold - mnew);
        sc_s[rowq] = sc;
        l_s[rowq]  = l_s[rowq] * sc + lloc;
        m_s[rowq]  = mnew;
      }
    }
    __syncthreads();

    // O = diag(sc) * O + P V    (K-dim = 64 -> 2 WMMA steps per tile)
    #pragma unroll
    for (int c = 0; c < 4; ++c) {
      #pragma unroll
      for (int r = 0; r < 8; ++r)
        o[c][r] *= sc_s[tm * 16 + r + 8 * hi];
      #pragma unroll
      for (int kk = 0; kk < 2; ++kk) {
        v16bf ap = *(const v16bf*)&SP[(tm * 16 + lr) * 64 + kk * 32 + lh];
        v16bf bv = *(const v16bf*)&Vs[((onb + c) * 16 + lr) * 64 + kk * 32 + lh];
        o[c] = wmma_bf16(ap, bv, o[c]);
      }
    }
  }

  // epilogue: /l, *nu, write [B,T,H*D]
  #pragma unroll
  for (int c = 0; c < 4; ++c)
    #pragma unroll
    for (int r = 0; r < 8; ++r) {
      int rowq = tm * 16 + r + 8 * hi;
      int d    = (onb + c) * 16 + lr;
      float val = o[c][r] / l_s[rowq] * nuv;
      out[((size_t)b * HT + (q0 + rowq)) * HC + h * HD + d] = val;
    }
}

// ---------------------------------------------------------------- launch
extern "C" void kernel_launch(void* const* d_in, const int* in_sizes, int n_in,
                              void* d_out, int out_size, void* d_ws, size_t ws_size,
                              hipStream_t stream) {
  const float* x   = (const float*)d_in[0];
  const float* wq  = (const float*)d_in[1];
  const float* wk  = (const float*)d_in[2];
  const float* wv  = (const float*)d_in[3];
  const float* tau = (const float*)d_in[4];
  const float* nu  = (const float*)d_in[5];
  float* out = (float*)d_out;

  const size_t NX = (size_t)HM * HC;   // 8,388,608
  const size_t NW = (size_t)HC * HC;   // 4,194,304
  char* ws = (char*)d_ws;
  u16* xb    = (u16*)ws;   ws += 2 * NX;
  u16* wqb   = (u16*)ws;   ws += 2 * NW;
  u16* wkb   = (u16*)ws;   ws += 2 * NW;
  u16* wvb   = (u16*)ws;   ws += 2 * NW;
  float* tmp = (float*)ws; ws += 4 * NX;
  u16* qb    = (u16*)ws;   ws += 2 * NX;
  u16* kb    = (u16*)ws;   ws += 2 * NX;
  u16* vb    = (u16*)ws;   ws += 2 * NX;   // total ~126 MB

  f32_to_bf16_kernel<<<(NX + 255) / 256, 256, 0, stream>>>(x,  xb,  (long)NX);
  f32_to_bf16_kernel<<<(NW + 255) / 256, 256, 0, stream>>>(wq, wqb, (long)NW);
  f32_to_bf16_kernel<<<(NW + 255) / 256, 256, 0, stream>>>(wk, wkb, (long)NW);
  f32_to_bf16_kernel<<<(NW + 255) / 256, 256, 0, stream>>>(wv, wvb, (long)NW);

  dim3 ggrid(HC / 128, HM / 128);   // (16, 32)
  // q
  gemm_nt_bf16<<<ggrid, 256, 0, stream>>>(xb, wqb, tmp, HM, HC, HC);
  l2norm_bf16_kernel<<<HM * HH, 128, 0, stream>>>(tmp, qb);
  // k (reuse tmp)
  gemm_nt_bf16<<<ggrid, 256, 0, stream>>>(xb, wkb, tmp, HM, HC, HC);
  l2norm_bf16_kernel<<<HM * HH, 128, 0, stream>>>(tmp, kb);
  // v
  gemm_nt_bf16<<<ggrid, 256, 0, stream>>>(xb, wvb, tmp, HM, HC, HC);
  l2norm_bf16_kernel<<<HM * HH, 128, 0, stream>>>(tmp, vb);

  attn_kernel<<<dim3(HT / 64, HB * HH), 256, 0, stream>>>(qb, kb, vb, out, tau, nu);
}
